// PhysicsAwareAttention_77403900609279
// MI455X (gfx1250) — compile-verified, matrix operand order checked by reading, not run on hardware
//
#include <hip/hip_runtime.h>
#include <stdint.h>

typedef __attribute__((ext_vector_type(16))) __bf16 v16bf;
typedef __attribute__((ext_vector_type(8)))  float  v8f;
typedef __attribute__((ext_vector_type(4)))  unsigned int u32x4;
typedef __attribute__((ext_vector_type(8)))  int i32x8;
typedef __attribute__((ext_vector_type(4)))  int i32x4;

constexpr int Bc = 2;
constexpr int Sc = 2048;
constexpr int Dc = 1024;
constexpr int Hc = 16;
constexpr int DKc = 64;
constexpr int SPc = 128;
constexpr int PDc = 3;

// ---------------------------------------------------------------------------
// WMMA helpers (CDNA5 wave32, 16x16x32 bf16 -> f32)
// ---------------------------------------------------------------------------
static __device__ __forceinline__ v8f wmma_bf16(v16bf a, v16bf b, v8f c) {
  return __builtin_amdgcn_wmma_f32_16x16x32_bf16(
      /*neg_a=*/false, a, /*neg_b=*/false, b,
      /*c_mod=*/(short)0, c, /*reuse_a=*/false, /*reuse_b=*/false);
}

// Load a 16x32 bf16 fragment from LDS per ISA 7.12.2:
// element (r, k) lives at tile[r*rs + k*ks]; lane L holds r = L%16,
// K set = {kb..kb+7} U {16+kb..16+kb+7} with kb = (L/16)*8, packed pairwise.
static __device__ __forceinline__ v16bf load_frag(const __bf16* tile, int rs, int ks) {
  const int lane = threadIdx.x & 31;
  const int r  = lane & 15;
  const int kb = (lane >> 4) << 3;
  v16bf f;
#pragma unroll
  for (int i = 0; i < 8; ++i) {
    const int k = (i < 4) ? (kb + 2 * i) : (16 + kb + 2 * (i - 4));
    f[2 * i]     = tile[r * rs + k * ks];
    f[2 * i + 1] = tile[r * rs + (k + 1) * ks];
  }
  return f;
}

// ---------------------------------------------------------------------------
// Tensor Data Mover: DMA a 2D bf16 tile (tile_d1 rows x tile_d0 cols, row
// stride stride0 elements) from global memory into LDS. D# per ISA ch.8.
// One instruction per issuing wave; tracked by TENSORcnt.
// This toolchain exposes the 6-arg builtin form.
// ---------------------------------------------------------------------------
static __device__ __forceinline__ unsigned lds_addr_of(const void* p) {
  return (unsigned)(unsigned long long)(uintptr_t)p;  // flat addr[31:0] = LDS offset
}

static __device__ __forceinline__ void tdm_load_2d(const __bf16* gsrc, __bf16* ldst,
                                                   unsigned tile_d0, unsigned tile_d1,
                                                   unsigned tensor_d0, unsigned tensor_d1,
                                                   unsigned stride0) {
  const unsigned long long ga = (unsigned long long)(uintptr_t)gsrc;
  u32x4 g0;
  g0[0] = 1u;                                        // count=1 (valid), user mode
  g0[1] = lds_addr_of(ldst);                         // lds_addr
  g0[2] = (unsigned)ga;                              // global_addr[31:0]
  g0[3] = (unsigned)((ga >> 32) & 0x01FFFFFFull)     // global_addr[56:32]
        | (2u << 30);                                // type = 2 ("image")
  i32x8 g1;
  g1[0] = (int)(1u << 16);                           // data_size = 1 -> 2 bytes
  g1[1] = (int)(tensor_d0 << 16);                    // tensor_dim0 lo16
  g1[2] = (int)((tensor_d0 >> 16) | (tensor_d1 << 16));  // dim0 hi16 | dim1 lo16
  g1[3] = (int)((tensor_d1 >> 16) | (tile_d0 << 16));    // dim1 hi16 | tile_dim0
  g1[4] = (int)tile_d1;                              // tile_dim1 | tile_dim2=0
  g1[5] = (int)stride0;                              // tensor_dim0_stride lo32
  g1[6] = 0;                                         // stride0 hi16 | dim1_stride lo16
  g1[7] = 0;
  const i32x4 z4 = {0, 0, 0, 0};                     // groups 2/3 unused (2D tensor)
  const i32x8 z8 = {0, 0, 0, 0, 0, 0, 0, 0};
  __builtin_amdgcn_tensor_load_to_lds(g0, g1, z4, z4, z8, 0);
}

// ---------------------------------------------------------------------------
// Fused QKV projection: out[b,h,s,dk] = (X @ W + bias)  (+ physics for K)
// mode 0: Q (scaled by 1/sqrt(DK))   mode 1: K (+ pb * pc)   mode 2: V
// ---------------------------------------------------------------------------
__global__ __launch_bounds__(256)
void proj_qkv(const float* __restrict__ X, const float* __restrict__ W,
              const float* __restrict__ bias, __bf16* __restrict__ out,
              int mode, const float* __restrict__ phys,
              const float* __restrict__ Pw, const float* __restrict__ pbp) {
  const int m0 = blockIdx.x * 128;
  const int n0 = blockIdx.y * 128;
  const int tid = threadIdx.x;
  const int wv = tid >> 5;
  const int wrow = (wv >> 1) * 32;   // 4x2 wave grid, 32x64 per wave
  const int wcol = (wv & 1) * 64;

  __shared__ __align__(16) __bf16 As[128 * 32];   // X tile, row-major [m][k]
  __shared__ __align__(16) __bf16 Bs[128 * 32];   // W tile transposed [n][k]

  v8f acc[2][4];
#pragma unroll
  for (int mt = 0; mt < 2; ++mt)
#pragma unroll
    for (int nt = 0; nt < 4; ++nt) acc[mt][nt] = {};

  for (int k0 = 0; k0 < Dc; k0 += 32) {
    __syncthreads();
    {  // X tile: 128x32 f32 -> bf16
      const int i  = tid >> 1;
      const int kk = (tid & 1) * 16;
      const float* src = X + (size_t)(m0 + i) * Dc + k0 + kk;
#pragma unroll
      for (int e = 0; e < 16; e += 4) {
        float4 v = *(const float4*)(src + e);
        As[i * 32 + kk + e + 0] = (__bf16)v.x;
        As[i * 32 + kk + e + 1] = (__bf16)v.y;
        As[i * 32 + kk + e + 2] = (__bf16)v.z;
        As[i * 32 + kk + e + 3] = (__bf16)v.w;
      }
    }
    {  // W tile: 32x128 f32 -> bf16, transposed into [n][k]
      const int kk = tid >> 3;
      const int nb = (tid & 7) * 16;
      const float* src = W + (size_t)(k0 + kk) * Dc + n0 + nb;
#pragma unroll
      for (int e = 0; e < 16; e += 4) {
        float4 v = *(const float4*)(src + e);
        Bs[(nb + e + 0) * 32 + kk] = (__bf16)v.x;
        Bs[(nb + e + 1) * 32 + kk] = (__bf16)v.y;
        Bs[(nb + e + 2) * 32 + kk] = (__bf16)v.z;
        Bs[(nb + e + 3) * 32 + kk] = (__bf16)v.w;
      }
    }
    __syncthreads();
#pragma unroll
    for (int mt = 0; mt < 2; ++mt) {
      v16bf a = load_frag(&As[(wrow + mt * 16) * 32], 32, 1);
#pragma unroll
      for (int nt = 0; nt < 4; ++nt) {
        v16bf b = load_frag(&Bs[(wcol + nt * 16) * 32], 32, 1);
        acc[mt][nt] = wmma_bf16(a, b, acc[mt][nt]);
      }
    }
  }

  // Epilogue: bias, mode-specific transform, scatter to [B,H,S,DK] bf16
  const int lane = tid & 31;
  const int nloc = lane & 15;
  const int mb = (lane >> 4) << 3;
  const float pbv = (mode == 1) ? pbp[0] : 0.f;
#pragma unroll
  for (int mt = 0; mt < 2; ++mt)
#pragma unroll
    for (int nt = 0; nt < 4; ++nt)
#pragma unroll
      for (int r = 0; r < 8; ++r) {
        const int m = m0 + wrow + mt * 16 + mb + r;
        const int n = n0 + wcol + nt * 16 + nloc;
        const int b = m >> 11;          // / Sc
        const int s = m & (Sc - 1);
        const int h = n >> 6;
        const int dk = n & 63;
        float v = acc[mt][nt][r] + bias[n];
        if (mode == 0) v *= 0.125f;     // 1/sqrt(64) folded into Q
        if (mode == 1) {
          const float* ph = phys + ((size_t)b * SPc + (s >> 4)) * PDc;
          const float* pw = Pw + ((s & 15) << 6) + dk;
          const float pc = ph[0] * pw[0] + ph[1] * pw[Dc] + ph[2] * pw[2 * Dc];
          v += pbv * pc;
        }
        out[(((size_t)b * Hc + h) * Sc + s) * DKc + dk] = (__bf16)v;
      }
}

// ---------------------------------------------------------------------------
// Flash attention: per (b,h), 128 q-rows per WG, 8 waves x 16 rows,
// TDM double-buffered 64-key K/V tiles, online softmax.
// Writes ctx as bf16 [B,S,D].
// ---------------------------------------------------------------------------
__global__ __launch_bounds__(256)
void flash_attn(const __bf16* __restrict__ Qh, const __bf16* __restrict__ Kh,
                const __bf16* __restrict__ Vh, __bf16* __restrict__ Ctx) {
  const int b = blockIdx.z, h = blockIdx.y;
  const int q0 = blockIdx.x * 128;
  const int tid = threadIdx.x;
  const int wv = tid >> 5;
  const int lane = tid & 31;
  const int nloc = lane & 15;
  const int mb = (lane >> 4) << 3;
  constexpr int NT = Sc / 64;

  __shared__ __align__(16) __bf16 Ks[2][64 * 64];     // [buf][key][dk]
  __shared__ __align__(16) __bf16 Vs[2][64 * 64];     // [buf][key][dk]
  __shared__ __align__(16) __bf16 Ps[8][16 * 64];     // per-wave P / Q staging

  const size_t headoff = ((size_t)b * Hc + h) * Sc * DKc;
  const __bf16* Kbase = Kh + headoff;
  const __bf16* Vbase = Vh + headoff;

  // Kick off TDM for tile 0 into buffer 0 (wave 0 only).
  if (wv == 0) {
    tdm_load_2d(Kbase, &Ks[0][0], 64, 64, 64, 64, 64);
    tdm_load_2d(Vbase, &Vs[0][0], 64, 64, 64, 64, 64);
  }

  // Stage this wave's 16x64 Q strip and build 2 A-fragments (k over DK).
  {
    const uint4* src = (const uint4*)(Qh + headoff + (size_t)(q0 + wv * 16) * DKc);
    uint4* dst = (uint4*)&Ps[wv][0];
#pragma unroll
    for (int j = 0; j < 4; ++j) dst[lane * 4 + j] = src[lane * 4 + j];
  }
  v16bf qf[2];
  qf[0] = load_frag(&Ps[wv][0], 64, 1);
  qf[1] = load_frag(&Ps[wv][32], 64, 1);

  v8f cacc[4];
#pragma unroll
  for (int dt = 0; dt < 4; ++dt) cacc[dt] = {};
  float mrun[8], lrun[8];
#pragma unroll
  for (int r = 0; r < 8; ++r) { mrun[r] = -1e30f; lrun[r] = 0.f; }

  for (int kt = 0; kt < NT; ++kt) {
    const int cur = kt & 1;
    // Prefetch tile kt+1 into the other buffer, then wait until tile kt's
    // two TDM ops are complete (tensor ops complete in order per wave).
    if (wv == 0) {
      if (kt + 1 < NT) {
        const __bf16* kn = Kbase + (size_t)(kt + 1) * 64 * DKc;
        const __bf16* vn = Vbase + (size_t)(kt + 1) * 64 * DKc;
        tdm_load_2d(kn, &Ks[cur ^ 1][0], 64, 64, 64, 64, 64);
        tdm_load_2d(vn, &Vs[cur ^ 1][0], 64, 64, 64, 64, 64);
        __builtin_amdgcn_s_wait_tensorcnt(2);
      } else {
        __builtin_amdgcn_s_wait_tensorcnt(0);
      }
    }
    __syncthreads();

    // scores: S[16x64] = Qstrip(16xDK) @ Ktile^T
    v8f sacc[4];
#pragma unroll
    for (int nt = 0; nt < 4; ++nt) {
      sacc[nt] = {};
#pragma unroll
      for (int ks2 = 0; ks2 < 2; ++ks2) {
        v16bf kb = load_frag(&Ks[cur][(nt * 16) * 64 + ks2 * 32], 64, 1);
        sacc[nt] = wmma_bf16(qf[ks2], kb, sacc[nt]);
      }
    }

    // online softmax: rows mb..mb+7 live in this lane's half
    float rmax[8];
#pragma unroll
    for (int r = 0; r < 8; ++r) {
      float v = sacc[0][r];
#pragma unroll
      for (int nt = 1; nt < 4; ++nt) v = fmaxf(v, sacc[nt][r]);
#pragma unroll
      for (int msk = 1; msk < 16; msk <<= 1) v = fmaxf(v, __shfl_xor(v, msk, 32));
      rmax[r] = v;
    }
    float mnew[8], scl[8], lad[8];
#pragma unroll
    for (int r = 0; r < 8; ++r) {
      mnew[r] = fmaxf(mrun[r], rmax[r]);
      scl[r] = __expf(mrun[r] - mnew[r]);
      lad[r] = 0.f;
    }
#pragma unroll
    for (int nt = 0; nt < 4; ++nt)
#pragma unroll
      for (int r = 0; r < 8; ++r) {
        const float p = __expf(sacc[nt][r] - mnew[r]);
        lad[r] += p;
        Ps[wv][(mb + r) * 64 + nt * 16 + nloc] = (__bf16)p;
      }
#pragma unroll
    for (int r = 0; r < 8; ++r) {
#pragma unroll
      for (int msk = 1; msk < 16; msk <<= 1) lad[r] += __shfl_xor(lad[r], msk, 32);
      lrun[r] = lrun[r] * scl[r] + lad[r];
      mrun[r] = mnew[r];
    }
#pragma unroll
    for (int dt = 0; dt < 4; ++dt)
#pragma unroll
      for (int r = 0; r < 8; ++r) cacc[dt][r] *= scl[r];

    // ctx += P(16x64) @ Vtile(64x64); V is k-major so k_stride = 64
#pragma unroll
    for (int ks2 = 0; ks2 < 2; ++ks2) {
      v16bf pf = load_frag(&Ps[wv][ks2 * 32], 64, 1);
#pragma unroll
      for (int dt = 0; dt < 4; ++dt) {
        v16bf vf = load_frag(&Vs[cur][(ks2 * 32) * 64 + dt * 16], 1, 64);
        cacc[dt] = wmma_bf16(pf, vf, cacc[dt]);
      }
    }
    __syncthreads();  // all consumers done before next TDM overwrites buffers
  }

  // normalize and write ctx in [B,S,D] bf16 layout
#pragma unroll
  for (int dt = 0; dt < 4; ++dt)
#pragma unroll
    for (int r = 0; r < 8; ++r) {
      const int row = q0 + wv * 16 + mb + r;
      const int dfull = h * 64 + dt * 16 + nloc;
      Ctx[((size_t)b * Sc + row) * Dc + dfull] = (__bf16)(cacc[dt][r] / lrun[r]);
    }
}

// ---------------------------------------------------------------------------
// Output projection: out[f32] = ctx[bf16] @ Wo + bo
// ---------------------------------------------------------------------------
__global__ __launch_bounds__(256)
void gemm_out(const __bf16* __restrict__ A, const float* __restrict__ W,
              const float* __restrict__ bias, float* __restrict__ out) {
  const int m0 = blockIdx.x * 128;
  const int n0 = blockIdx.y * 128;
  const int tid = threadIdx.x;
  const int wv = tid >> 5;
  const int wrow = (wv >> 1) * 32;
  const int wcol = (wv & 1) * 64;

  __shared__ __align__(16) __bf16 As[128 * 32];
  __shared__ __align__(16) __bf16 Bs[128 * 32];

  v8f acc[2][4];
#pragma unroll
  for (int mt = 0; mt < 2; ++mt)
#pragma unroll
    for (int nt = 0; nt < 4; ++nt) acc[mt][nt] = {};

  for (int k0 = 0; k0 < Dc; k0 += 32) {
    __syncthreads();
    {  // A tile already bf16: straight b128 copies
      const int i  = tid >> 1;
      const int kk = (tid & 1) * 16;
      const uint4* src = (const uint4*)(A + (size_t)(m0 + i) * Dc + k0 + kk);
      uint4* dst = (uint4*)&As[i * 32 + kk];
      dst[0] = src[0];
      dst[1] = src[1];
    }
    {
      const int kk = tid >> 3;
      const int nb = (tid & 7) * 16;
      const float* src = W + (size_t)(k0 + kk) * Dc + n0 + nb;
#pragma unroll
      for (int e = 0; e < 16; e += 4) {
        float4 v = *(const float4*)(src + e);
        Bs[(nb + e + 0) * 32 + kk] = (__bf16)v.x;
        Bs[(nb + e + 1) * 32 + kk] = (__bf16)v.y;
        Bs[(nb + e + 2) * 32 + kk] = (__bf16)v.z;
        Bs[(nb + e + 3) * 32 + kk] = (__bf16)v.w;
      }
    }
    __syncthreads();
#pragma unroll
    for (int mt = 0; mt < 2; ++mt) {
      v16bf a = load_frag(&As[(wrow + mt * 16) * 32], 32, 1);
#pragma unroll
      for (int nt = 0; nt < 4; ++nt) {
        v16bf b = load_frag(&Bs[(wcol + nt * 16) * 32], 32, 1);
        acc[mt][nt] = wmma_bf16(a, b, acc[mt][nt]);
      }
    }
  }

  const int lane = tid & 31;
  const int nloc = lane & 15;
  const int mb = (lane >> 4) << 3;
#pragma unroll
  for (int mt = 0; mt < 2; ++mt)
#pragma unroll
    for (int nt = 0; nt < 4; ++nt)
#pragma unroll
      for (int r = 0; r < 8; ++r) {
        const int m = m0 + wrow + mt * 16 + mb + r;
        const int n = n0 + wcol + nt * 16 + nloc;
        out[(size_t)m * Dc + n] = acc[mt][nt][r] + bias[n];
      }
}

// ---------------------------------------------------------------------------
extern "C" void kernel_launch(void* const* d_in, const int* in_sizes, int n_in,
                              void* d_out, int out_size, void* d_ws, size_t ws_size,
                              hipStream_t stream) {
  const float* query  = (const float*)d_in[0];
  const float* key_in = (const float*)d_in[1];
  const float* value  = (const float*)d_in[2];
  const float* phys   = (const float*)d_in[3];
  const float* Wq = (const float*)d_in[4];
  const float* bq = (const float*)d_in[5];
  const float* Wk = (const float*)d_in[6];
  const float* bk = (const float*)d_in[7];
  const float* Wv = (const float*)d_in[8];
  const float* bv = (const float*)d_in[9];
  const float* Wo = (const float*)d_in[10];
  const float* bo = (const float*)d_in[11];
  const float* Pw = (const float*)d_in[12];
  const float* pb = (const float*)d_in[13];

  const size_t per = (size_t)Bc * Hc * Sc * DKc;  // 4M elements -> 8MB bf16
  __bf16* Qh  = (__bf16*)d_ws;
  __bf16* Kh  = Qh + per;
  __bf16* Vh  = Kh + per;
  __bf16* Ctx = Vh + per;

  dim3 blk(256);
  dim3 gp(Bc * Sc / 128, Dc / 128);  // (32, 8)

  proj_qkv<<<gp, blk, 0, stream>>>(query,  Wq, bq, Qh, 0, nullptr, nullptr, nullptr);
  proj_qkv<<<gp, blk, 0, stream>>>(key_in, Wk, bk, Kh, 1, phys, Pw, pb);
  proj_qkv<<<gp, blk, 0, stream>>>(value,  Wv, bv, Vh, 2, nullptr, nullptr, nullptr);

  flash_attn<<<dim3(Sc / 128, Hc, Bc), blk, 0, stream>>>(Qh, Kh, Vh, Ctx);

  gemm_out<<<gp, blk, 0, stream>>>(Ctx, Wo, bo, (float*)d_out);
}
